// MultiHeadAttention_5205500363312
// MI455X (gfx1250) — matrix operand order repaired and checked by reading, not compiled, not run on hardware
//
#include <hip/hip_runtime.h>
#include <stdint.h>

typedef __attribute__((ext_vector_type(16))) __bf16 v16bf;
typedef __attribute__((ext_vector_type(8)))  __bf16 v8bf;
typedef __attribute__((ext_vector_type(8)))  float  v8f;
typedef __attribute__((ext_vector_type(4)))  float  v4f;

#define INPUT_DIM 1024
#define HEADS     16
#define DK        64
#define BATCH     4
#define SEQ       2048
#define MROWS     (BATCH*SEQ)   /* 8192 */

#define LOG2E     1.4426950408889634f
#define NEGBIG    (-1.4426950e9f)      /* -1e9 in exp2 domain */

static __device__ __forceinline__ v16bf cat16(v8bf lo, v8bf hi) {
  return __builtin_shufflevector(lo, hi, 0,1,2,3,4,5,6,7,8,9,10,11,12,13,14,15);
}
static __device__ __forceinline__ v8f wmma_bf16(v16bf a, v16bf b, v8f c) {
  // D = A(16x32 bf16) * B(32x16 bf16) + C(16x16 f32)
  return __builtin_amdgcn_wmma_f32_16x16x32_bf16(false, a, false, b, (short)0, c,
                                                 false, false);
}
static __device__ __forceinline__ float fast_exp2(float x) {
  return __builtin_amdgcn_exp2f(x);    // raw v_exp_f32
}

// ---------------------------------------------------------------------------
// GEMM: C[M,N] = ((A[M,K] * W[N,K]^T) + bias[N]) * outScale
// Software-pipelined: global loads for tile i+1 issue before compute of tile i.
// Block: 256 threads (8 waves, 4(M) x 2(N)); block tile 128x128;
// wave tile 32(M) x 64(N) = 2x4 WMMA tiles; K-step 32.
// ---------------------------------------------------------------------------
template<typename AT, bool OUT_F32>
__global__ __launch_bounds__(256)
void gemm_xt(const AT* __restrict__ A, const float* __restrict__ W,
             const float* __restrict__ bias, void* __restrict__ Cout,
             int M, int N, int K, float outScale)
{
  constexpr int AS = 40;   // LDS row stride (elements): 80B, 16B-aligned, spreads banks
  constexpr int BS = 40;
  __shared__ __bf16 Asm[128 * AS];
  __shared__ __bf16 Bsm[128 * BS];

  const int t    = threadIdx.x;
  const int lane = t & 31, w = t >> 5;
  const int wm   = (w & 3) * 32;
  const int wn   = (w >> 2) * 64;
  const int bm   = blockIdx.y * 128;
  const int bn   = blockIdx.x * 128;
  const int half = lane >> 4, l16 = lane & 15;

  // staging registers (next tile in flight)
  v4f  aRf[4];            // float-A path
  v8bf aRb[2];            // bf16-A path
  v4f  bR[4];

  auto loadTiles = [&](int kb) {
    if constexpr (sizeof(AT) == 4) {
      const float* Af = (const float*)A;
      #pragma unroll
      for (int i = 0; i < 4; ++i) {
        int c = t + i * 256;                 // 1024 float4 chunks (128x32)
        int row = c >> 3, kc = (c & 7) << 2;
        aRf[i] = *(const v4f*)(Af + (size_t)(bm + row) * K + kb + kc);
      }
    } else {
      const __bf16* Ab = (const __bf16*)A;
      #pragma unroll
      for (int i = 0; i < 2; ++i) {
        int c = t + i * 256;                 // 512 v8bf chunks
        int row = c >> 2, kc = (c & 3) << 3;
        aRb[i] = *(const v8bf*)(Ab + (size_t)(bm + row) * K + kb + kc);
      }
    }
    #pragma unroll
    for (int i = 0; i < 4; ++i) {
      int c = t + i * 256;                   // 1024 float4 chunks (128x32)
      int row = c >> 3, kc = (c & 7) << 2;
      bR[i] = *(const v4f*)(W + (size_t)(bn + row) * K + kb + kc);
    }
  };
  auto storeTiles = [&]() {
    if constexpr (sizeof(AT) == 4) {
      #pragma unroll
      for (int i = 0; i < 4; ++i) {
        int c = t + i * 256;
        int row = c >> 3, kc = (c & 7) << 2;
        __bf16* dst = &Asm[row * AS + kc];
        dst[0] = (__bf16)aRf[i][0]; dst[1] = (__bf16)aRf[i][1];
        dst[2] = (__bf16)aRf[i][2]; dst[3] = (__bf16)aRf[i][3];
      }
    } else {
      #pragma unroll
      for (int i = 0; i < 2; ++i) {
        int c = t + i * 256;
        int row = c >> 2, kc = (c & 3) << 3;
        *(v8bf*)&Asm[row * AS + kc] = aRb[i];
      }
    }
    #pragma unroll
    for (int i = 0; i < 4; ++i) {
      int c = t + i * 256;
      int row = c >> 3, kc = (c & 7) << 2;
      __bf16* dst = &Bsm[row * BS + kc];
      dst[0] = (__bf16)bR[i][0]; dst[1] = (__bf16)bR[i][1];
      dst[2] = (__bf16)bR[i][2]; dst[3] = (__bf16)bR[i][3];
    }
  };

  v8f acc[2][4] = {};

  loadTiles(0);
  storeTiles();
  __syncthreads();

  for (int kb = 0; kb < K; kb += 32) {
    const bool hasNext = (kb + 32 < K);
    if (hasNext) {
      loadTiles(kb + 32);                    // overlap HBM/L2 latency with WMMA
      if (kb + 64 < K)                       // prefetch two tiles ahead
        __builtin_prefetch((const void*)(A + (size_t)(bm + (t >> 1)) * K + kb + 64), 0, 1);
    }

    v16bf af[2], bf2[4];
    #pragma unroll
    for (int mt = 0; mt < 2; ++mt) {
      const __bf16* ap = &Asm[(wm + mt * 16 + l16) * AS];
      af[mt] = cat16(*(const v8bf*)(ap + half * 8),
                     *(const v8bf*)(ap + 16 + half * 8));
    }
    #pragma unroll
    for (int nt = 0; nt < 4; ++nt) {
      const __bf16* bp = &Bsm[(wn + nt * 16 + l16) * BS];
      bf2[nt] = cat16(*(const v8bf*)(bp + half * 8), *(const v8bf*)(bp + 16 + half * 8));
    }
    #pragma unroll
    for (int mt = 0; mt < 2; ++mt)
      #pragma unroll
      for (int nt = 0; nt < 4; ++nt)
        acc[mt][nt] = wmma_bf16(af[mt], bf2[nt], acc[mt][nt]);

    __syncthreads();                         // done reading this LDS tile
    if (hasNext) {
      storeTiles();
      __syncthreads();
    }
  }

  // ---- epilogue: (acc + bias) * outScale ----
  #pragma unroll
  for (int nt = 0; nt < 4; ++nt) {
    int col = bn + wn + nt * 16 + l16;
    float bv = bias[col];
    #pragma unroll
    for (int mt = 0; mt < 2; ++mt) {
      #pragma unroll
      for (int v = 0; v < 8; ++v) {
        int row = bm + wm + mt * 16 + v + 8 * half;
        float val = (acc[mt][nt][v] + bv) * outScale;
        if constexpr (OUT_F32)
          ((float*)Cout)[(size_t)row * N + col] = val;
        else
          ((__bf16*)Cout)[(size_t)row * N + col] = (__bf16)val;
      }
    }
  }
}

// ---------------------------------------------------------------------------
// Flash attention: one workgroup (4 waves / 128 thr) per (b, h, 64 q-rows).
// Q pre-scaled by log2(e)/sqrt(dk); softmax runs in exp2 domain (raw v_exp).
// Mask staged per-chunk into LDS as additive bias (0 / -1e9*log2e).
// Software-pipelined: next chunk's global loads issue before current compute.
// Row sums of P via WMMA against an all-ones B fragment.
// ---------------------------------------------------------------------------
__global__ __launch_bounds__(128)
void flash_attn(const __bf16* __restrict__ Qg, const __bf16* __restrict__ Kg,
                const __bf16* __restrict__ Vg, const int* __restrict__ mask,
                __bf16* __restrict__ Ctx)
{
  constexpr int KS = 72;   // Ksm [key][d]   32x64 bf16
  constexpr int VS = 40;   // Vsm [d][key]   64x32 bf16 (transposed)
  constexpr int PS = 40;   // P staging per wave 16x32 bf16
  constexpr int MS = 68;   // Msm [key][qrow] 32x64 f32 additive bias
  __shared__ __bf16 Ksm[32 * KS];
  __shared__ __bf16 Vsm[64 * VS];
  __shared__ __bf16 Psm[4 * 16 * PS];
  __shared__ float  Msm[32 * MS];

  const int t = threadIdx.x, lane = t & 31, wave = t >> 5;
  const int half = lane >> 4, l16 = lane & 15;
  const int qblk = blockIdx.x, bh = blockIdx.y;
  const int b = bh >> 4, h = bh & 15;           // HEADS == 16
  const int qrow0 = qblk * 64 + wave * 16;
  const size_t baseBS = (size_t)b * SEQ;

  // Q fragments for this wave's 16 rows (kept in VGPRs for whole kernel)
  v16bf aq[2];
  {
    const __bf16* qp = Qg + (baseBS + qrow0 + l16) * INPUT_DIM + h * DK;
    #pragma unroll
    for (int kw = 0; kw < 2; ++kw)
      aq[kw] = cat16(*(const v8bf*)(qp + kw * 32 + half * 8),
                     *(const v8bf*)(qp + kw * 32 + 16 + half * 8));
  }

  v16bf onesb;                                  // all-ones B frag for row sums
  #pragma unroll
  for (int e = 0; e < 16; ++e) onesb[e] = (__bf16)1.0f;

  float mrun[8], lrun[8];
  v8f acc[4] = {};
  #pragma unroll
  for (int v = 0; v < 8; ++v) { mrun[v] = -1e30f; lrun[v] = 0.0f; }

  __bf16* Pw = &Psm[wave * 16 * PS];
  const int rbase = wave * 16 + 8 * half;

  // staging registers for the chunk in flight
  v8bf kR[2], vR[2];
  int4 mR[4];

  auto loadChunk = [&](int kcN) {
    #pragma unroll
    for (int i = 0; i < 2; ++i) {
      int c = t + i * 128;                 // 256 v8bf chunks (32x64)
      int key = c >> 3, dc = (c & 7) << 3;
      kR[i] = *(const v8bf*)(Kg + (baseBS + kcN + key) * INPUT_DIM + h * DK + dc);
      vR[i] = *(const v8bf*)(Vg + (baseBS + kcN + key) * INPUT_DIM + h * DK + dc);
    }
    const int* mbase = mask + (baseBS + qblk * 64) * SEQ + kcN;
    #pragma unroll
    for (int i = 0; i < 4; ++i) {
      int c = t + i * 128;                 // 512 int4 chunks (64 rows x 32 keys)
      int qr = c >> 3, kcc = (c & 7) << 2;
      mR[i] = *(const int4*)(mbase + (size_t)qr * SEQ + kcc);
    }
  };
  auto storeChunk = [&]() {
    #pragma unroll
    for (int i = 0; i < 2; ++i) {
      int c = t + i * 128;
      int key = c >> 3, dc = (c & 7) << 3;
      *(v8bf*)&Ksm[key * KS + dc] = kR[i];
      #pragma unroll
      for (int j = 0; j < 8; ++j) Vsm[(dc + j) * VS + key] = vR[i][j];
    }
    #pragma unroll
    for (int i = 0; i < 4; ++i) {
      int c = t + i * 128;
      int qr = c >> 3, kcc = (c & 7) << 2;
      Msm[(kcc + 0) * MS + qr] = mR[i].x ? 0.0f : NEGBIG;
      Msm[(kcc + 1) * MS + qr] = mR[i].y ? 0.0f : NEGBIG;
      Msm[(kcc + 2) * MS + qr] = mR[i].z ? 0.0f : NEGBIG;
      Msm[(kcc + 3) * MS + qr] = mR[i].w ? 0.0f : NEGBIG;
    }
  };

  loadChunk(0);
  storeChunk();
  __syncthreads();

  for (int kc = 0; kc < SEQ; kc += 32) {
    const bool hasNext = (kc + 32 < SEQ);
    if (hasNext) loadChunk(kc + 32);       // overlap with compute below

    // ---- mask bias (issue LDS reads early) ----
    v4f m0a = *(const v4f*)&Msm[l16 * MS + rbase];
    v4f m0b = *(const v4f*)&Msm[l16 * MS + rbase + 4];
    v4f m1a = *(const v4f*)&Msm[(16 + l16) * MS + rbase];
    v4f m1b = *(const v4f*)&Msm[(16 + l16) * MS + rbase + 4];

    // ---- scores: S(16q x 32k), 2 key-subtiles x 2 d-windows ----
    v8f s0 = {}, s1 = {};
    #pragma unroll
    for (int kw = 0; kw < 2; ++kw) {
      const __bf16* bp0 = &Ksm[l16 * KS + kw * 32];
      v16bf bk0 = cat16(*(const v8bf*)(bp0 + half * 8), *(const v8bf*)(bp0 + 16 + half * 8));
      s0 = wmma_bf16(aq[kw], bk0, s0);
      const __bf16* bp1 = &Ksm[(16 + l16) * KS + kw * 32];
      v16bf bk1 = cat16(*(const v8bf*)(bp1 + half * 8), *(const v8bf*)(bp1 + 16 + half * 8));
      s1 = wmma_bf16(aq[kw], bk1, s1);
    }
    #pragma unroll
    for (int v = 0; v < 4; ++v) {
      s0[v]     += m0a[v];  s0[v + 4] += m0b[v];
      s1[v]     += m1a[v];  s1[v + 4] += m1b[v];
    }

    // ---- online softmax (exp2 domain): max via shfl tree ----
    float p0[8], p1[8], corr[8];
    #pragma unroll
    for (int v = 0; v < 8; ++v) {
      float tm = fmaxf(s0[v], s1[v]);
      #pragma unroll
      for (int off = 1; off < 16; off <<= 1)
        tm = fmaxf(tm, __shfl_xor(tm, off, 16));
      float mn = fmaxf(mrun[v], tm);
      corr[v]  = fast_exp2(mrun[v] - mn);
      mrun[v]  = mn;
      p0[v] = fast_exp2(s0[v] - mn);
      p1[v] = fast_exp2(s1[v] - mn);
      #pragma unroll
      for (int dt = 0; dt < 4; ++dt) acc[dt][v] *= corr[v];
    }

    // ---- bounce P through LDS: C-layout -> A-fragment layout (wave-local) ----
    #pragma unroll
    for (int v = 0; v < 8; ++v) {
      int r = v + 8 * half;
      Pw[r * PS + l16]      = (__bf16)p0[v];
      Pw[r * PS + 16 + l16] = (__bf16)p1[v];
    }
    v16bf ap;
    {
      const __bf16* pp = &Pw[l16 * PS];
      ap = cat16(*(const v8bf*)(pp + half * 8), *(const v8bf*)(pp + 16 + half * 8));
    }

    // ---- row sums of P via matrix pipe: rs = P @ ones ----
    {
      v8f rs = {};
      rs = wmma_bf16(ap, onesb, rs);
      #pragma unroll
      for (int v = 0; v < 8; ++v) lrun[v] = lrun[v] * corr[v] + rs[v];
    }

    // ---- ctx(16x64) += P(16x32) @ V(32x64) ----
    #pragma unroll
    for (int dt = 0; dt < 4; ++dt) {
      const __bf16* vp = &Vsm[(dt * 16 + l16) * VS];
      v16bf bv = cat16(*(const v8bf*)(vp + half * 8), *(const v8bf*)(vp + 16 + half * 8));
      acc[dt] = wmma_bf16(ap, bv, acc[dt]);
    }

    __syncthreads();                       // done reading this chunk's LDS
    if (hasNext) {
      storeChunk();
      __syncthreads();
    }
  }

  // ---- normalize + store context (bf16, [b, s, h*64+d]) ----
  #pragma unroll
  for (int v = 0; v < 8; ++v) {
    float linv = 1.0f / lrun[v];
    int row = qrow0 + v + 8 * half;
    #pragma unroll
    for (int dt = 0; dt < 4; ++dt)
      Ctx[(baseBS + row) * INPUT_DIM + h * DK + dt * 16 + l16] =
          (__bf16)(acc[dt][v] * linv);
  }
}

// ---------------------------------------------------------------------------
extern "C" void kernel_launch(void* const* d_in, const int* in_sizes, int n_in,
                              void* d_out, int out_size, void* d_ws, size_t ws_size,
                              hipStream_t stream)
{
  (void)in_sizes; (void)n_in; (void)out_size; (void)ws_size;
  const float* q    = (const float*)d_in[0];
  const float* k    = (const float*)d_in[1];
  const float* v    = (const float*)d_in[2];
  const int*   mask = (const int*)  d_in[3];
  const float* Wq   = (const float*)d_in[4];
  const float* bq   = (const float*)d_in[5];
  const float* Wk   = (const float*)d_in[6];
  const float* bk   = (const float*)d_in[7];
  const float* Wv   = (const float*)d_in[8];
  const float* bv   = (const float*)d_in[9];
  const float* Wo   = (const float*)d_in[10];
  const float* bo   = (const float*)d_in[11];
  float* out = (float*)d_out;

  const size_t MK = (size_t)MROWS * INPUT_DIM;       // 8M elems per tensor
  __bf16* Qb = (__bf16*)d_ws;                        // 16 MB each, 64 MB total
  __bf16* Kb = Qb + MK;
  __bf16* Vb = Kb + MK;
  __bf16* Cx = Vb + MK;

  dim3 gg(INPUT_DIM / 128, MROWS / 128);             // (8, 64)
  // Q pre-scaled by log2(e)/sqrt(DK) so the softmax runs on raw v_exp2
  gemm_xt<float,  false><<<gg, 256, 0, stream>>>(q, Wq, bq, Qb, MROWS, INPUT_DIM, INPUT_DIM,
                                                 0.125f * LOG2E);
  gemm_xt<float,  false><<<gg, 256, 0, stream>>>(k, Wk, bk, Kb, MROWS, INPUT_DIM, INPUT_DIM, 1.0f);
  gemm_xt<float,  false><<<gg, 256, 0, stream>>>(v, Wv, bv, Vb, MROWS, INPUT_DIM, INPUT_DIM, 1.0f);

  dim3 gf(SEQ / 64, BATCH * HEADS);                  // (32, 64)
  flash_attn<<<gf, 128, 0, stream>>>(Qb, Kb, Vb, mask, Cx);

  gemm_xt<__bf16, true><<<gg, 256, 0, stream>>>(Cx, Wo, bo, out, MROWS, INPUT_DIM, INPUT_DIM, 1.0f);
}
